// st_attention_59691455479830
// MI455X (gfx1250) — compile-verified, matrix operand order checked by reading, not run on hardware
//
#include <hip/hip_runtime.h>
#include <hip/hip_fp16.h>

typedef __attribute__((ext_vector_type(16))) _Float16 h16;
typedef __attribute__((ext_vector_type(8)))  _Float16 h8;
typedef __attribute__((ext_vector_type(8)))  float    f8;

#define NV    25
#define NH    3
#define CIN   64
#define COUT  64
#define RELD  8
#define NB    128
#define TT    128
#define TTILE 16
#define TWN   (TTILE * NV)   // 400 (t,v) columns per tile
#define BN_EPS 1e-5f

// Assemble a 16-half WMMA operand from two 16-byte chunks (ISA 7.12.2:
// halves 0..7 = K base .. base+7, halves 8..15 = K base+16 .. base+23).
static __device__ inline h16 ld_frag(const _Float16* p) {
    union { h16 v; h8 h[2]; } u;
    u.h[0] = *(const h8*)(p);
    u.h[1] = *(const h8*)(p + 16);
    return u.v;
}

// ---------------- kernel 0: zero BN partial sums ----------------
__global__ void k_init(float* sums, float* sumsq) {
    int i = threadIdx.x;
    if (i < COUT) { sums[i] = 0.f; sumsq[i] = 0.f; }
}

// ---------------- kernel 1: xm = mean over T ----------------
__global__ __launch_bounds__(128) void k_mean(const float* __restrict__ x,
                                              float* __restrict__ xm) {
    __shared__ float s[NV];
    int nc  = blockIdx.x;            // n*64 + c
    int tid = threadIdx.x;           // 128 threads
    if (tid < NV) s[tid] = 0.f;
    __syncthreads();
    const float* p = x + (size_t)nc * TT * NV;
    for (int chunk = 0; chunk < 25; ++chunk) {
        int l = chunk * 128 + tid;   // 0..3199, coalesced
        atomicAdd(&s[l % NV], p[l]);
    }
    __syncthreads();
    if (tid < NV) xm[nc * NV + tid] = s[tid] * (1.f / (float)TT);
}

// ---------------- kernel 2: k,q -> tanh scores -> FF -> attB (f16, B-layout) ----
// attB[h][n][o][v(32)][w(32)], zero-padded so WMMA K/N padding is exact.
__global__ __launch_bounds__(256) void k_att(
    const float* __restrict__ xm,
    const float* __restrict__ WKw, const float* __restrict__ WKb,
    const float* __restrict__ WQw, const float* __restrict__ WQb,
    const float* __restrict__ FFw, const float* __restrict__ FFb,
    const float* __restrict__ dpa_a, const float* __restrict__ dpa_b,
    const float* __restrict__ blk_a, const float* __restrict__ adjc,
    _Float16* __restrict__ attB) {
    __shared__ float sxm[CIN * NV];          // 6400 B
    __shared__ float sk[RELD * NV];          // 800 B
    __shared__ float sq[RELD * NV];          // 800 B
    __shared__ float ss[RELD * NV * NV];     // 20000 B
    int h = blockIdx.x / NB, n = blockIdx.x % NB;
    int tid = threadIdx.x;

    for (int i = tid; i < CIN * NV; i += 256) sxm[i] = xm[n * CIN * NV + i];
    __syncthreads();

    for (int i = tid; i < RELD * NV; i += 256) {
        int r = i / NV, v = i % NV;
        float a0 = WKb[h * RELD + r], a1 = WQb[h * RELD + r];
        for (int c = 0; c < CIN; ++c) {
            float xv = sxm[c * NV + v];
            a0 += xv * WKw[(h * RELD + r) * CIN + c];
            a1 += xv * WQw[(h * RELD + r) * CIN + c];
        }
        sk[i] = a0; sq[i] = a1;
    }
    __syncthreads();

    float al = dpa_a[h], be = dpa_b[h];
    for (int i = tid; i < RELD * NV * NV; i += 256) {
        int r = i / (NV * NV), vw = i % (NV * NV);
        int v = vw / NV, w = vw % NV;
        float kk = sk[r * NV + v], qq = sq[r * NV + w];
        ss[i] = tanhf(al * kk * qq + be * (kk - qq));
    }
    __syncthreads();

    float ba = blk_a[h];
    _Float16* dst = attB + (size_t)(h * NB + n) * COUT * 1024;
    for (int i = tid; i < COUT * 1024; i += 256) {
        int o = i >> 10, rem = i & 1023, v = rem >> 5, w = rem & 31;
        float val = 0.f;
        if (v < NV && w < NV) {
            float f = FFb[h * COUT + o];
#pragma unroll
            for (int r = 0; r < RELD; ++r)
                f += FFw[(h * COUT + o) * RELD + r] * ss[r * NV * NV + v * NV + w];
            val = ba * f + adjc[(h * NV + v) * NV + w];
        }
        dst[i] = (_Float16)val;
    }
}

// ---------------- kernel 3: main WMMA kernel ----------------
// One block per (n, 16-wide t-tile). LDS (273 KB of the WGP's 320 KB):
//   xl  [400][64]  f16  (x chunk, transposed so K=c is lane-contiguous)
//   vl  [3][64][16][32] f16 (all 3 heads' value tiles -> single live accumulator
//                            in GEMM2, h is a K-accumulation loop)
//   wvl [3][64][64] f16 + bias
__global__ __launch_bounds__(256) void k_main(
    const float* __restrict__ x, const float* __restrict__ WVw,
    const float* __restrict__ WVb, const _Float16* __restrict__ attB,
    float* __restrict__ out_pre, float* __restrict__ sums,
    float* __restrict__ sumsq) {
    extern __shared__ char smem[];
    _Float16* xl  = (_Float16*)smem;                 // 400*64   = 51200 B
    _Float16* vl  = xl + TWN * CIN;                  // 3*64*16*32 = 196608 B
    _Float16* wvl = vl + NH * COUT * TTILE * 32;     // 3*64*64  = 24576 B
    float*   wvbl = (float*)(wvl + NH * COUT * CIN); // 192*4    = 768 B

    int n = blockIdx.x >> 3, t0 = (blockIdx.x & 7) * TTILE;
    int tid = threadIdx.x, lane = tid & 31;
    // Force wave-uniformity so tile loops lower to scalar branches
    // (EXEC stays all-ones around every WMMA, per ISA 7.12 restriction).
    int wave = __builtin_amdgcn_readfirstlane(tid >> 5);
    int g = lane >> 4, lrow = lane & 15;

    // stage WV weights (f32 -> f16) and bias
    for (int i = tid; i < NH * COUT * CIN; i += 256) wvl[i] = (_Float16)WVw[i];
    for (int i = tid; i < NH * COUT; i += 256) wvbl[i] = WVb[i];
    // stage x chunk transposed: xl[tw][c]  (coalesced global reads)
    const float* xp = x + (size_t)n * CIN * TT * NV + t0 * NV;
    for (int i = tid; i < CIN * TWN; i += 256) {
        int c = i / TWN, tw = i % TWN;
        xl[tw * CIN + c] = (_Float16)xp[c * TT * NV + tw];
    }
    // warm L2/WGP$ with this wave's GEMM2 B-operand slice (global_prefetch_b8)
    {
        const _Float16* pb = attB + (size_t)n * COUT * 1024 + (wave * 8) * 1024;
        for (int h = 0; h < NH; ++h)
            __builtin_prefetch(pb + (size_t)h * NB * COUT * 1024 + lane * 256, 0, 1);
    }
    __syncthreads();

    // GEMM1: v_h[o, tw] = WV_h(64x64) * x(64x400) for all 3 heads -> LDS
    for (int tIdx = wave; tIdx < NH * 100; tIdx += 8) {
        int h  = tIdx / 100, rem = tIdx % 100;
        int mt = rem / 25,   nt  = rem % 25;
        f8 acc = {};
#pragma unroll
        for (int ks = 0; ks < 2; ++ks) {
            h16 A = ld_frag(wvl + (h * COUT + mt * 16 + lrow) * CIN + ks * 32 + g * 8);
            h16 B = ld_frag(xl + (nt * 16 + lrow) * CIN + ks * 32 + g * 8);
            acc = __builtin_amdgcn_wmma_f32_16x16x32_f16(false, A, false, B,
                                                         (short)0, acc, false, false);
        }
        int tw = nt * 16 + lrow, t = tw / NV, w = tw % NV;
        _Float16* vdst = vl + h * COUT * TTILE * 32;
#pragma unroll
        for (int r = 0; r < 8; ++r) {
            int o = mt * 16 + r + 8 * g;
            vdst[(o * TTILE + t) * 32 + w] = (_Float16)(acc[r] + wvbl[h * COUT + o]);
        }
    }
    __syncthreads();

    // GEMM2: out[t,v] = sum_h sum_w v_h[t,w] * att_h[v,w]   (K = w, padded 25->32)
    const _Float16* ab = attB + (size_t)n * COUT * 1024;
#pragma unroll
    for (int i = 0; i < 8; ++i) {
        int o = wave * 8 + i;
        float s1 = 0.f, s2 = 0.f;
#pragma unroll
        for (int vt = 0; vt < 2; ++vt) {
            f8 acc = {};
#pragma unroll
            for (int h = 0; h < NH; ++h) {
                h16 A = ld_frag(vl + (h * COUT + o) * TTILE * 32 + lrow * 32 + g * 8);
                h16 B = ld_frag(ab + (size_t)h * NB * COUT * 1024 + o * 1024
                                   + (vt * 16 + lrow) * 32 + g * 8);
                acc = __builtin_amdgcn_wmma_f32_16x16x32_f16(false, A, false, B,
                                                             (short)0, acc, false, false);
            }
            int v = vt * 16 + lrow;
#pragma unroll
            for (int r = 0; r < 8; ++r) {
                int t = r + 8 * g;
                float val = acc[r];
                if (v < NV) {
                    out_pre[(((size_t)n * COUT + o) * TT + t0 + t) * NV + v] = val;
                    s1 += val; s2 += val * val;
                }
            }
        }
        // wave-level reduction for BN partials, then 2 atomics per (block, o)
        for (int off = 16; off > 0; off >>= 1) {
            s1 += __shfl_xor(s1, off, 32);
            s2 += __shfl_xor(s2, off, 32);
        }
        if (lane == 0) { atomicAdd(&sums[o], s1); atomicAdd(&sumsq[o], s2); }
    }
}

// ---------------- kernel 4: finalize BN stats ----------------
__global__ void k_stats(const float* sums, const float* sumsq, float* mu, float* rs) {
    int o = threadIdx.x;
    if (o < COUT) {
        float cnt = (float)NB * (float)TT * (float)NV;
        float m = sums[o] / cnt;
        float var = sumsq[o] / cnt - m * m;
        mu[o] = m;
        rs[o] = rsqrtf(var + BN_EPS);
    }
}

// ---------------- kernel 5: BN apply + residual + relu ----------------
__global__ __launch_bounds__(256) void k_final(
    const float* __restrict__ pre, const float* __restrict__ x,
    const float* __restrict__ mu, const float* __restrict__ rs,
    const float* __restrict__ gamma, const float* __restrict__ beta,
    float* __restrict__ out) {
    size_t i4 = (size_t)blockIdx.x * 256 + threadIdx.x;
    size_t base = i4 * 4;
    int o = (int)((base / (TT * NV)) % COUT);
    float4 p  = *(const float4*)(pre + base);
    float4 xv = *(const float4*)(x + base);
    float m = mu[o], r = rs[o], gg = gamma[o], bb = beta[o];
    float4 res;
    res.x = fmaxf((p.x - m) * r * gg + bb + xv.x, 0.f);
    res.y = fmaxf((p.y - m) * r * gg + bb + xv.y, 0.f);
    res.z = fmaxf((p.z - m) * r * gg + bb + xv.z, 0.f);
    res.w = fmaxf((p.w - m) * r * gg + bb + xv.w, 0.f);
    *(float4*)(out + base) = res;
}

extern "C" void kernel_launch(void* const* d_in, const int* in_sizes, int n_in,
                              void* d_out, int out_size, void* d_ws, size_t ws_size,
                              hipStream_t stream) {
    const float* x     = (const float*)d_in[0];
    const float* WKw   = (const float*)d_in[1];
    const float* WKb   = (const float*)d_in[2];
    const float* WQw   = (const float*)d_in[3];
    const float* WQb   = (const float*)d_in[4];
    const float* WVw   = (const float*)d_in[5];
    const float* WVb   = (const float*)d_in[6];
    const float* FFw   = (const float*)d_in[7];
    const float* FFb   = (const float*)d_in[8];
    const float* dpa_a = (const float*)d_in[9];
    const float* dpa_b = (const float*)d_in[10];
    const float* blk_a = (const float*)d_in[11];
    const float* adjc  = (const float*)d_in[12];
    const float* gam   = (const float*)d_in[13];
    const float* bet   = (const float*)d_in[14];
    float* out = (float*)d_out;

    char* ws = (char*)d_ws;
    float*     xm    = (float*)ws;                       // 819,200 B
    float*     sums  = (float*)(ws + 1048576);           // 64 f32
    float*     sumsq = sums + 64;
    float*     mu    = sums + 128;
    float*     rs    = sums + 192;
    _Float16*  attB  = (_Float16*)(ws + 2 * 1048576);    // 50,331,648 B
    float*     pre   = (float*)(ws + 2 * 1048576 + 50331648); // 104,857,600 B

    k_init<<<1, 128, 0, stream>>>(sums, sumsq);
    k_mean<<<NB * CIN, 128, 0, stream>>>(x, xm);
    k_att<<<NH * NB, 256, 0, stream>>>(xm, WKw, WKb, WQw, WQb, FFw, FFb,
                                       dpa_a, dpa_b, blk_a, adjc, attB);
    size_t shmem = (size_t)TWN * CIN * 2              // xl
                 + (size_t)NH * COUT * TTILE * 32 * 2 // vl
                 + (size_t)NH * COUT * CIN * 2        // wvl
                 + (size_t)NH * COUT * 4;             // wvbl  => 273,152 B
    k_main<<<NB * (TT / TTILE), 256, shmem, stream>>>(x, WVw, WVb, attB, pre,
                                                      sums, sumsq);
    k_stats<<<1, 64, 0, stream>>>(sums, sumsq, mu, rs);
    size_t total = (size_t)NB * COUT * TT * NV;       // 26,214,400
    k_final<<<(unsigned)(total / 4 / 256), 256, 0, stream>>>(pre, x, mu, rs,
                                                             gam, bet, out);
}
